// TaskConditionedSharedMoE_27625229648260
// MI455X (gfx1250) — matrix operand order repaired and checked by reading, not compiled, not run on hardware
//
#include <hip/hip_runtime.h>
#include <hip/hip_bf16.h>
#include <math.h>

// ---------------------------------------------------------------------------
// TaskConditionedSharedMoE for MI455X (gfx1250, wave32, WMMA).
// All GEMMs run on v_wmma_f32_16x16x32_bf16 (fp32 accumulate).
// Prepass converts activations to bf16 (row-major) and weights to bf16
// transposed [N][K], so A/B fragments are bare global_load_b128 streams with
// no in-loop conversion VALU. Gate tiles staged to LDS with
// global_load_async_to_lds_b128 (ASYNCcnt).
// ---------------------------------------------------------------------------

typedef __attribute__((ext_vector_type(16))) __bf16 v16bf;
typedef __attribute__((ext_vector_type(8)))  __bf16 v8bf;
typedef __attribute__((ext_vector_type(4)))  __bf16 v4bf;
typedef __attribute__((ext_vector_type(8)))  float  v8f;

#define B_ROWS   32768
#define NOBS     480
#define NPRIV    768
#define NA       32
#define NE       8
#define NH       512
#define NT       64
#define GATEH    256
#define BT       128           // batch rows per workgroup
#define H1S      520           // padded LDS stride for H1 (bf16 elems)

union Bf16x16 { v16bf v; v8bf h[2]; };

static __device__ inline v8f wmma_bf16(v16bf a, v16bf b, v8f c) {
  return __builtin_amdgcn_wmma_f32_16x16x32_bf16(
      /*neg_a=*/false, a, /*neg_b=*/false, b,
      /*c_mod=*/(short)0, c, /*reuse_a=*/false, /*reuse_b=*/false);
}

static __device__ inline float eluf(float x) {
  return x > 0.f ? x : (__expf(x) - 1.f);
}

// A-fragment (16x32 bf16) from a row-major bf16 row pointer (global or LDS).
// lane<16 holds K {k0..k0+15}; lane>=16 holds {k0+16..k0+31}.
static __device__ inline v16bf a_frag(const __bf16* p, int k0, int hi) {
  Bf16x16 r;
  r.h[0] = *(const v8bf*)(p + k0 + hi * 16);
  r.h[1] = *(const v8bf*)(p + k0 + hi * 16 + 8);
  return r.v;
}

// B-fragment (32x16 bf16) from transposed bf16 weights WT[N][K].
static __device__ inline v16bf b_from_wt(const __bf16* __restrict__ wt, int col,
                                         int K, int k0, int hi) {
  return *(const v16bf*)(wt + (size_t)col * K + k0 + hi * 16);
}

// ---------------------------------------------------------------------------
// Prepass: fp32 [b][K][N] -> bf16 [b][N][K]  (weights)
// ---------------------------------------------------------------------------
__global__ void k_tr_bf16(const float* __restrict__ src, __bf16* __restrict__ dst,
                          int K, int N) {
  const float* s = src + (size_t)blockIdx.y * K * N;
  __bf16*      d = dst + (size_t)blockIdx.y * K * N;
  size_t total = (size_t)K * N;
  for (size_t i = (size_t)blockIdx.x * blockDim.x + threadIdx.x; i < total;
       i += (size_t)gridDim.x * blockDim.x) {
    int n = (int)(i / K);
    int k = (int)(i % K);
    d[(size_t)n * K + k] = (__bf16)s[(size_t)k * N + n];
  }
}

// Prepass: fp32 -> bf16, same layout (activations). n must be multiple of 4.
__global__ void k_cvt_bf16(const float* __restrict__ src, __bf16* __restrict__ dst,
                           long n) {
  long stride = (long)gridDim.x * blockDim.x * 4;
  for (long i = ((long)blockIdx.x * blockDim.x + threadIdx.x) * 4; i < n; i += stride) {
    float4 x = *(const float4*)(src + i);
    v4bf o;
    o[0] = (__bf16)x.x; o[1] = (__bf16)x.y; o[2] = (__bf16)x.z; o[3] = (__bf16)x.w;
    *(v4bf*)(dst + i) = o;
  }
}

// ---------------------------------------------------------------------------
// Gate: task = ELU(obs @ taskW + tb); gh = ELU(task @ gW1 + gb1);
//       gate = softmax(gh @ gW2 + gb2)
// One 128-row block, 8 waves (one 16-row stripe each), fused in LDS.
// ---------------------------------------------------------------------------
__global__ __launch_bounds__(256) void k_gate(
    const __bf16* __restrict__ X, const __bf16* __restrict__ tWT,
    const float* __restrict__ tb, const __bf16* __restrict__ g1T,
    const float* __restrict__ g1b, const __bf16* __restrict__ g2T,
    const float* __restrict__ g2b, float* __restrict__ gate_out,
    float* __restrict__ gate_ws) {
  extern __shared__ char smem[];
  __bf16* TE = (__bf16*)smem;                        // [128][72]
  __bf16* GH = (__bf16*)(smem + 128 * 72 * 2);       // [128][264]
  float*  LG = (float*)(smem + 128 * 72 * 2 + 128 * 264 * 2);  // [128][8]

  const int lane = threadIdx.x & 31;
  const int w    = threadIdx.x >> 5;
  const int l15  = lane & 15, hi = lane >> 4;
  const int rowLoc = w * 16;
  const __bf16* xrow = X + (size_t)(blockIdx.x * BT + rowLoc + l15) * NOBS;

  // --- task embed: [128x480] @ [480x64] ---
  {
    v8f acc[4] = {};
    for (int k0 = 0; k0 < NOBS; k0 += 32) {
      v16bf a = a_frag(xrow, k0, hi);
#pragma unroll
      for (int nt = 0; nt < 4; ++nt)
        acc[nt] = wmma_bf16(a, b_from_wt(tWT, nt * 16 + l15, NOBS, k0, hi), acc[nt]);
    }
#pragma unroll
    for (int nt = 0; nt < 4; ++nt) {
      int col = nt * 16 + l15;
      float bias = tb[col];
#pragma unroll
      for (int v = 0; v < 8; ++v)
        TE[(rowLoc + v + 8 * hi) * 72 + col] = (__bf16)eluf(acc[nt][v] + bias);
    }
  }
  asm volatile("s_wait_dscnt 0" ::: "memory");

  // --- gate hidden: [128x64] @ [64x256] ---
  for (int c = 0; c < 2; ++c) {
    v8f acc[8] = {};
#pragma unroll
    for (int k0 = 0; k0 < NT; k0 += 32) {
      v16bf a = a_frag(TE + (rowLoc + l15) * 72, k0, hi);
#pragma unroll
      for (int nt = 0; nt < 8; ++nt) {
        int col = c * 128 + nt * 16 + l15;
        acc[nt] = wmma_bf16(a, b_from_wt(g1T, col, NT, k0, hi), acc[nt]);
      }
    }
#pragma unroll
    for (int nt = 0; nt < 8; ++nt) {
      int col = c * 128 + nt * 16 + l15;
      float bias = g1b[col];
#pragma unroll
      for (int v = 0; v < 8; ++v)
        GH[(rowLoc + v + 8 * hi) * 264 + col] = (__bf16)eluf(acc[nt][v] + bias);
    }
  }
  asm volatile("s_wait_dscnt 0" ::: "memory");

  // --- logits: [128x256] @ [256x8] (padded to 16 cols) ---
  {
    v8f acc = {};
    for (int k0 = 0; k0 < GATEH; k0 += 32) {
      v16bf a = a_frag(GH + (rowLoc + l15) * 264, k0, hi);
      v16bf b = {};
      if (l15 < NE) b = b_from_wt(g2T, l15, GATEH, k0, hi);
      acc = wmma_bf16(a, b, acc);
    }
    if (l15 < NE) {
      float bias = g2b[l15];
#pragma unroll
      for (int v = 0; v < 8; ++v)
        LG[(rowLoc + v + 8 * hi) * NE + l15] = acc[v] + bias;
    }
  }
  __syncthreads();

  // --- per-row softmax over 8 experts ---
  if (threadIdx.x < BT) {
    int r = blockIdx.x * BT + threadIdx.x;
    float lg[NE], m = -1e30f, s = 0.f;
#pragma unroll
    for (int i = 0; i < NE; ++i) { lg[i] = LG[threadIdx.x * NE + i]; m = fmaxf(m, lg[i]); }
#pragma unroll
    for (int i = 0; i < NE; ++i) { lg[i] = __expf(lg[i] - m); s += lg[i]; }
    float inv = 1.f / s;
#pragma unroll
    for (int i = 0; i < NE; ++i) {
      float gw = lg[i] * inv;
      gate_out[(size_t)r * NE + i] = gw;
      gate_ws[(size_t)r * NE + i]  = gw;
    }
  }
}

// ---------------------------------------------------------------------------
// MoE branch (actor: DIN=480, NOUT=32 / critic: DIN=768, NOUT=1).
// Per 128-row block: loop experts; GEMM1 -> H1 (LDS bf16);
// GEMM2 fused with GEMM3 through a per-wave 16x64 bf16 scratch so H2 never
// leaves the WGP; gate tile staged via async-load-to-LDS; gate-weighted
// accumulation in registers.
// ---------------------------------------------------------------------------
__global__ __launch_bounds__(256) void k_moe(
    const __bf16* __restrict__ X, int DIN,
    const __bf16* __restrict__ W1T, const float* __restrict__ B1,
    const __bf16* __restrict__ W2T, const float* __restrict__ B2,
    const __bf16* __restrict__ W3T, const float* __restrict__ B3,
    const float* __restrict__ gate, float* __restrict__ out, int NOUT) {
  extern __shared__ char smem[];
  __bf16* H1 = (__bf16*)smem;                              // [128][H1S]
  __bf16* SC = (__bf16*)(smem + BT * H1S * 2);             // 8 waves x [16][72]
  float*  GL = (float*)(smem + BT * H1S * 2 + 8 * 16 * 72 * 2);  // [128][8]

  const int lane = threadIdx.x & 31;
  const int w    = threadIdx.x >> 5;
  const int l15  = lane & 15, hi = lane >> 4;
  const int rowLoc  = w * 16;
  const int rowBase = blockIdx.x * BT + rowLoc;
  const __bf16* xrow = X + (size_t)(rowBase + l15) * DIN;
  __bf16* sc = SC + w * 16 * 72;

  // Stage this block's gate tile [128][8] fp32 into LDS asynchronously.
  // Thread t copies 16B; wave w covers exactly rows 16w..16w+15 (its own),
  // so s_wait_asynccnt 0 in this wave suffices (no cross-wave barrier).
  {
    unsigned ldsOff = (unsigned)(uintptr_t)GL + (unsigned)threadIdx.x * 16u;
    unsigned long long gaddr =
        (unsigned long long)(uintptr_t)(gate + (size_t)blockIdx.x * BT * NE +
                                        (size_t)threadIdx.x * 4);
    asm volatile("global_load_async_to_lds_b128 %0, %1, off"
                 :: "v"(ldsOff), "v"(gaddr) : "memory");
  }

  v8f finalAcc[2] = {};

  for (int e = 0; e < NE; ++e) {
    const __bf16* w1 = W1T + (size_t)e * NH * DIN;
    const __bf16* w2 = W2T + (size_t)e * NH * NH;
    const __bf16* w3 = W3T + (size_t)e * NOUT * NH;
    if (e + 1 < NE)  // warm L2/WGP$ for the next expert's first-layer weights
      __builtin_prefetch(W1T + (size_t)(e + 1) * NH * DIN, 0, 1);

    // ---- GEMM1: H1 = ELU(X @ W1 + b1), 512 cols in 4 chunks of 128 ----
    for (int nc = 0; nc < 4; ++nc) {
      v8f acc[8] = {};
      for (int k0 = 0; k0 < DIN; k0 += 32) {
        v16bf a = a_frag(xrow, k0, hi);
#pragma unroll
        for (int nt = 0; nt < 8; ++nt) {
          int col = nc * 128 + nt * 16 + l15;
          acc[nt] = wmma_bf16(a, b_from_wt(w1, col, DIN, k0, hi), acc[nt]);
        }
      }
#pragma unroll
      for (int nt = 0; nt < 8; ++nt) {
        int col = nc * 128 + nt * 16 + l15;
        float bias = B1[e * NH + col];
#pragma unroll
        for (int v = 0; v < 8; ++v)
          H1[(rowLoc + v + 8 * hi) * H1S + col] = (__bf16)eluf(acc[nt][v] + bias);
      }
    }
    asm volatile("s_wait_dscnt 0" ::: "memory");

    // ---- GEMM2+GEMM3 fused over 64-col chunks of H2 ----
    v8f outacc[2] = {};
    for (int c = 0; c < 8; ++c) {
      const int n0 = c * 64;
      v8f h2[4] = {};
      for (int k0 = 0; k0 < NH; k0 += 32) {
        v16bf a = a_frag(H1 + (rowLoc + l15) * H1S, k0, hi);
#pragma unroll
        for (int nt = 0; nt < 4; ++nt) {
          int col = n0 + nt * 16 + l15;
          h2[nt] = wmma_bf16(a, b_from_wt(w2, col, NH, k0, hi), h2[nt]);
        }
      }
#pragma unroll
      for (int nt = 0; nt < 4; ++nt) {
        int col = n0 + nt * 16 + l15;
        float bias = B2[e * NH + col];
#pragma unroll
        for (int v = 0; v < 8; ++v)
          sc[(v + 8 * hi) * 72 + nt * 16 + l15] = (__bf16)eluf(h2[nt][v] + bias);
      }
      asm volatile("s_wait_dscnt 0" ::: "memory");
#pragma unroll
      for (int kk = 0; kk < 64; kk += 32) {
        v16bf a = a_frag(sc + l15 * 72, kk, hi);
#pragma unroll
        for (int ot = 0; ot < 2; ++ot) {
          int col = ot * 16 + l15;
          v16bf b = {};
          if (col < NOUT) b = *(const v16bf*)(w3 + (size_t)col * NH + n0 + kk + hi * 16);
          outacc[ot] = wmma_bf16(a, b, outacc[ot]);
        }
      }
    }

    // ---- gate-weighted accumulate (+ per-expert bias) ----
    asm volatile("s_wait_asynccnt 0" ::: "memory");
#pragma unroll
    for (int v = 0; v < 8; ++v) {
      float g = GL[(rowLoc + v + 8 * hi) * NE + e];
#pragma unroll
      for (int ot = 0; ot < 2; ++ot) {
        int col = ot * 16 + l15;
        float bias = (col < NOUT) ? B3[e * NOUT + col] : 0.f;
        finalAcc[ot][v] += g * (outacc[ot][v] + bias);
      }
    }
  }

  // ---- store ----
#pragma unroll
  for (int v = 0; v < 8; ++v) {
    int r = rowBase + v + 8 * hi;
#pragma unroll
    for (int ot = 0; ot < 2; ++ot) {
      int col = ot * 16 + l15;
      if (col < NOUT) out[(size_t)r * NOUT + col] = finalAcc[ot][v];
    }
  }
}

// ---------------------------------------------------------------------------
extern "C" void kernel_launch(void* const* d_in, const int* in_sizes, int n_in,
                              void* d_out, int out_size, void* d_ws, size_t ws_size,
                              hipStream_t stream) {
  (void)in_sizes; (void)n_in; (void)out_size; (void)ws_size;
  const float* obs   = (const float*)d_in[0];
  const float* priv  = (const float*)d_in[1];
  const float* taskW = (const float*)d_in[2];
  const float* taskb = (const float*)d_in[3];
  const float* gW1   = (const float*)d_in[4];
  const float* gb1   = (const float*)d_in[5];
  const float* gW2   = (const float*)d_in[6];
  const float* gb2   = (const float*)d_in[7];
  const float* aW1 = (const float*)d_in[8];  const float* ab1 = (const float*)d_in[9];
  const float* aW2 = (const float*)d_in[10]; const float* ab2 = (const float*)d_in[11];
  const float* aW3 = (const float*)d_in[12]; const float* ab3 = (const float*)d_in[13];
  const float* cW1 = (const float*)d_in[14]; const float* cb1 = (const float*)d_in[15];
  const float* cW2 = (const float*)d_in[16]; const float* cb2 = (const float*)d_in[17];
  const float* cW3 = (const float*)d_in[18]; const float* cb3 = (const float*)d_in[19];

  // Workspace layout: bf16 activations, bf16 transposed weights, fp32 gate.
  char* ws = (char*)d_ws;
  size_t off = 0;
  auto carve = [&](size_t bytes) { char* p = ws + off; off += (bytes + 255) & ~(size_t)255; return p; };
  __bf16* obsB  = (__bf16*)carve((size_t)B_ROWS * NOBS * 2);
  __bf16* privB = (__bf16*)carve((size_t)B_ROWS * NPRIV * 2);
  __bf16* tWT  = (__bf16*)carve((size_t)NT * NOBS * 2);
  __bf16* g1T  = (__bf16*)carve((size_t)GATEH * NT * 2);
  __bf16* g2T  = (__bf16*)carve((size_t)NE * GATEH * 2);
  __bf16* aW1T = (__bf16*)carve((size_t)NE * NH * NOBS * 2);
  __bf16* aW2T = (__bf16*)carve((size_t)NE * NH * NH * 2);
  __bf16* aW3T = (__bf16*)carve((size_t)NE * NA * NH * 2);
  __bf16* cW1T = (__bf16*)carve((size_t)NE * NH * NPRIV * 2);
  __bf16* cW2T = (__bf16*)carve((size_t)NE * NH * NH * 2);
  __bf16* cW3T = (__bf16*)carve((size_t)NE * 1 * NH * 2);
  float*  gateWs = (float*)carve((size_t)B_ROWS * NE * 4);

  // Prepass: activations fp32 -> bf16 (same layout)
  k_cvt_bf16<<<4096, 256, 0, stream>>>(obs,  obsB,  (long)B_ROWS * NOBS);
  k_cvt_bf16<<<4096, 256, 0, stream>>>(priv, privB, (long)B_ROWS * NPRIV);

  // Prepass: weights fp32 [K][N] -> bf16 [N][K]
  auto tr = [&](const float* s, __bf16* d, int K, int N, int batch) {
    dim3 g(512, batch, 1);
    k_tr_bf16<<<g, 256, 0, stream>>>(s, d, K, N);
  };
  tr(taskW, tWT, NOBS, NT, 1);
  tr(gW1,   g1T, NT, GATEH, 1);
  tr(gW2,   g2T, GATEH, NE, 1);
  tr(aW1, aW1T, NOBS, NH, NE);
  tr(aW2, aW2T, NH, NH, NE);
  tr(aW3, aW3T, NH, NA, NE);
  tr(cW1, cW1T, NPRIV, NH, NE);
  tr(cW2, cW2T, NH, NH, NE);
  tr(cW3, cW3T, NH, 1, NE);

  float* actorOut  = (float*)d_out;                       // [B,32]
  float* criticOut = actorOut + (size_t)B_ROWS * NA;      // [B,1]
  float* gateOut   = criticOut + (size_t)B_ROWS;          // [B,8]

  const int nBlocks = B_ROWS / BT;                        // 256
  const size_t gateSh = 128 * 72 * 2 + 128 * 264 * 2 + 128 * NE * 4;          // ~88 KB
  const size_t moeSh  = (size_t)BT * H1S * 2 + 8 * 16 * 72 * 2 + BT * NE * 4; // ~152 KB

  (void)hipFuncSetAttribute((const void*)k_gate,
      hipFuncAttributeMaxDynamicSharedMemorySize, (int)gateSh);
  (void)hipFuncSetAttribute((const void*)k_moe,
      hipFuncAttributeMaxDynamicSharedMemorySize, (int)moeSh);

  k_gate<<<nBlocks, 256, gateSh, stream>>>(obsB, tWT, taskb, g1T, gb1, g2T, gb2,
                                           gateOut, gateWs);
  k_moe<<<nBlocks, 256, moeSh, stream>>>(obsB, NOBS, aW1T, ab1, aW2T, ab2,
                                         aW3T, ab3, gateWs, actorOut, NA);
  k_moe<<<nBlocks, 256, moeSh, stream>>>(privB, NPRIV, cW1T, cb1, cW2T, cb2,
                                         cW3T, cb3, gateWs, criticOut, 1);
}
